// MultiHeadCausalAttention_37452114821627
// MI455X (gfx1250) — matrix operand order repaired and checked by reading, not compiled, not run on hardware
//
#include <hip/hip_runtime.h>
#include <hip/hip_bf16.h>

typedef __attribute__((ext_vector_type(16))) __bf16       v16bf;
typedef __attribute__((ext_vector_type(8)))  float        v8f;
typedef __attribute__((ext_vector_type(4)))  unsigned int u32x4;
typedef __attribute__((ext_vector_type(4)))  float        f32x4;
typedef unsigned short u16;

union Frag { u32x4 q[2]; v16bf v; };

__device__ __forceinline__ u16 f2bf(float f) {
  unsigned u = __float_as_uint(f);
  u += 0x7FFFu + ((u >> 16) & 1u);   // round-to-nearest-even
  return (u16)(u >> 16);
}

__device__ __forceinline__ v8f vzero() {
  v8f z;
  for (int i = 0; i < 8; i++) z[i] = 0.f;
  return z;
}

// fp32 -> bf16 conversion, grid-stride
__global__ void cvt_bf16(const float* __restrict__ in, u16* __restrict__ out, int n) {
  int i = blockIdx.x * blockDim.x + threadIdx.x;
  int stride = gridDim.x * blockDim.x;
  for (; i < n; i += stride) out[i] = f2bf(in[i]);
}

// ---------------------------------------------------------------------------
// Tiled bf16 GEMM: 128x128 block tile, 8 waves, 32x64 wave tile (2x4 frags),
// fp32 accumulate. 8 WMMAs per 6 fragment loads per wave per K32-step.
// Fragment conventions (CDNA5 ISA 7.12.2):
//   A frag : lane -> M = l&15 ; element e -> K = (e>>3)*16 + (l>>4)*8 + (e&7)
//   B frag : lane -> N = l&15 ; element e -> K = (e>>3)*16 + (l>>4)*8 + (e&7)
//   C/D    : lane -> N = l&15 ; VGPR i -> M = i + 8*(l>>4)
// mode 0: bf16 output scattered to (B,H,S,hd) layout (QKV projections)
// mode 1: fp32 output + bias, row-major (final projection)
// ---------------------------------------------------------------------------
__global__ __launch_bounds__(256) void gemm128_bf16(
    const void* __restrict__ Aptr, int a_fp32,
    const u16* __restrict__ Bw,
    int M, int N, int K, int S, int H,
    u16* __restrict__ out_bf, float* __restrict__ out_f,
    const float* __restrict__ bias, int mode)
{
  __shared__ u16 As[128 * 40];   // [m][k]  stride 40 halves (80B rows)
  __shared__ u16 Bs[32 * 136];   // [k][n]  stride 136 halves (272B rows)
  const int tid = threadIdx.x;
  const int wv = tid >> 5, l = tid & 31, hl = l >> 4, ln = l & 15;
  const int wm = wv >> 1, wn = wv & 1;           // 4x2 waves -> 32x64 each
  const int m0 = blockIdx.y * 128, n0 = blockIdx.x * 128;

  v8f acc[2][4];
  for (int si = 0; si < 2; si++)
    for (int sj = 0; sj < 4; sj++) acc[si][sj] = vzero();

  for (int k0 = 0; k0 < K; k0 += 32) {
    __syncthreads();
    { // stage A tile 128x32 (convert fp32->bf16 on the fly for x)
      int idx = tid * 16;
      int r = idx >> 5, cc = idx & 31;
      if (a_fp32) {
        const f32x4* g = (const f32x4*)((const float*)Aptr + (size_t)(m0 + r) * K + k0 + cc);
        float buf[16];
        #pragma unroll
        for (int j = 0; j < 4; j++) *(f32x4*)(buf + 4 * j) = g[j];
        #pragma unroll
        for (int j = 0; j < 16; j++) As[r * 40 + cc + j] = f2bf(buf[j]);
      } else {
        const u32x4* g = (const u32x4*)((const u16*)Aptr + (size_t)(m0 + r) * K + k0 + cc);
        *(u32x4*)&As[r * 40 + cc]     = g[0];
        *(u32x4*)&As[r * 40 + cc + 8] = g[1];
      }
    }
    { // stage B tile 32x128 row-major (coalesced both sides)
      int kk = tid >> 3, nn = (tid & 7) * 16;
      const u32x4* g = (const u32x4*)(Bw + (size_t)(k0 + kk) * N + n0 + nn);
      *(u32x4*)&Bs[kk * 136 + nn]     = g[0];
      *(u32x4*)&Bs[kk * 136 + nn + 8] = g[1];
    }
    __syncthreads();

    Frag fa[2], fb[4];
    #pragma unroll
    for (int si = 0; si < 2; si++) {
      const u16* p = &As[(wm * 32 + si * 16 + ln) * 40 + hl * 8];
      fa[si].q[0] = *(const u32x4*)p;
      fa[si].q[1] = *(const u32x4*)(p + 16);
    }
    #pragma unroll
    for (int sj = 0; sj < 4; sj++) {
      const u16* p = &Bs[wn * 64 + sj * 16 + ln];
      #pragma unroll
      for (int s = 0; s < 16; s++) ((u16*)&fb[sj])[s] = p[((s >> 3) * 16 + hl * 8 + (s & 7)) * 136];
    }
    #pragma unroll
    for (int si = 0; si < 2; si++)
      #pragma unroll
      for (int sj = 0; sj < 4; sj++)
        acc[si][sj] = __builtin_amdgcn_wmma_f32_16x16x32_bf16(
            false, fa[si].v, false, fb[sj].v, (short)0, acc[si][sj], false, false);
  }

  #pragma unroll
  for (int si = 0; si < 2; si++)
    #pragma unroll
    for (int sj = 0; sj < 4; sj++)
      #pragma unroll
      for (int i = 0; i < 8; i++) {
        int r = m0 + wm * 32 + si * 16 + i + hl * 8;
        int c = n0 + wn * 64 + sj * 16 + ln;
        float vval = acc[si][sj][i];
        if (mode == 0) {
          int b = r / S, s = r - b * S;
          size_t oidx = (((size_t)b * H + (c >> 6)) * S + s) * 64 + (c & 63);
          out_bf[oidx] = f2bf(vval);
        } else {
          out_f[(size_t)r * N + c] = vval + bias[c];
        }
      }
}

// ---------------------------------------------------------------------------
// Flash attention: one 64-row Q tile of one (b,h) per block; 4 waves, each
// owns 16 query rows. Online softmax in fp32; P goes through LDS to flip
// from C/D layout (lane->N) to A layout (lane->M) for the PV matmul.
// ---------------------------------------------------------------------------
__global__ __launch_bounds__(128) void attn64(
    const u16* __restrict__ Q, const u16* __restrict__ Km, const u16* __restrict__ Vm,
    u16* __restrict__ O, int S, int H)
{
  __shared__ u16 Kt[64 * 72];        // [hd k][key n]  (transposed K tile)
  __shared__ u16 Vs[64 * 72];        // [key k][hd n]  (row-major V tile)
  __shared__ u16 Ps[4][16 * 72];     // per-wave P tile [row][key]
  const int qt = blockIdx.x, h = blockIdx.y, b = blockIdx.z;
  const int tid = threadIdx.x, w = tid >> 5, l = tid & 31, hl = l >> 4, ln = l & 15;
  const size_t hoff = ((size_t)b * H + h) * (size_t)S * 64;
  const u16* Qh = Q + hoff; const u16* Kh = Km + hoff; const u16* Vh = Vm + hoff;
  const int qbase = qt * 64;
  const int qrow = qbase + w * 16 + ln;
  const float scale = 0.03125f;       // 1/sqrt(1024): module scales by d_out

  Frag fq[2];
  #pragma unroll
  for (int c = 0; c < 2; c++) {       // A-frag of Q straight from global
    const u16* p = Qh + (size_t)qrow * 64 + c * 32 + hl * 8;
    fq[c].q[0] = *(const u32x4*)p;
    fq[c].q[1] = *(const u32x4*)(p + 16);
  }

  v8f o[4];
  float mrow[8], lrow[8];
  #pragma unroll
  for (int nt = 0; nt < 4; nt++) o[nt] = vzero();
  #pragma unroll
  for (int i = 0; i < 8; i++) { mrow[i] = -__builtin_inff(); lrow[i] = 0.f; }

  for (int kt = 0; kt <= qt; kt++) {
    const int kb = kt * 64;
    __syncthreads();
    { // stage K (transposed) and V (row-major)
      int r = tid >> 1, c0 = (tid & 1) * 32;
      const u32x4* gk = (const u32x4*)(Kh + (size_t)(kb + r) * 64 + c0);
      u16 tmp[32];
      #pragma unroll
      for (int j = 0; j < 4; j++) *(u32x4*)(tmp + 8 * j) = gk[j];
      #pragma unroll
      for (int j = 0; j < 32; j++) Kt[(c0 + j) * 72 + r] = tmp[j];
      const u32x4* gv = (const u32x4*)(Vh + (size_t)(kb + r) * 64 + c0);
      #pragma unroll
      for (int j = 0; j < 4; j++) *(u32x4*)&Vs[r * 72 + c0 + 8 * j] = gv[j];
    }
    __syncthreads();

    // scores S = Q * K^T  (4 column tiles of 16 keys, K-dim = 64 in 2 chunks)
    v8f sc[4];
    #pragma unroll
    for (int t = 0; t < 4; t++) {
      sc[t] = vzero();
      #pragma unroll
      for (int c = 0; c < 2; c++) {
        Frag fk;
        const u16* p = &Kt[(c * 32) * 72 + t * 16 + ln];
        #pragma unroll
        for (int s = 0; s < 16; s++) ((u16*)&fk)[s] = p[((s >> 3) * 16 + hl * 8 + (s & 7)) * 72];
        sc[t] = __builtin_amdgcn_wmma_f32_16x16x32_bf16(
            false, fq[c].v, false, fk.v, (short)0, sc[t], false, false);
      }
      #pragma unroll
      for (int i = 0; i < 8; i++) sc[t][i] *= scale;
    }
    if (kt == qt) {   // causal mask only needed on diagonal block
      #pragma unroll
      for (int t = 0; t < 4; t++)
        #pragma unroll
        for (int i = 0; i < 8; i++) {
          int col = kb + t * 16 + ln;
          int row = qbase + w * 16 + i + hl * 8;
          if (col > row) sc[t][i] = -__builtin_inff();
        }
    }

    // online softmax: row stats live replicated across the 16-lane half
    float nm[8];
    #pragma unroll
    for (int i = 0; i < 8; i++) nm[i] = mrow[i];
    #pragma unroll
    for (int t = 0; t < 4; t++)
      #pragma unroll
      for (int i = 0; i < 8; i++) nm[i] = fmaxf(nm[i], sc[t][i]);
    #pragma unroll
    for (int mask = 1; mask < 16; mask <<= 1)
      #pragma unroll
      for (int i = 0; i < 8; i++) nm[i] = fmaxf(nm[i], __shfl_xor(nm[i], mask, 32));

    float alpha[8], rs[8];
    #pragma unroll
    for (int i = 0; i < 8; i++) {
      alpha[i] = __expf(mrow[i] - nm[i]);
      mrow[i] = nm[i];
      rs[i] = 0.f;
    }
    #pragma unroll
    for (int t = 0; t < 4; t++)
      #pragma unroll
      for (int i = 0; i < 8; i++) {
        float pv = __expf(sc[t][i] - mrow[i]);
        rs[i] += pv;
        Ps[w][(i + 8 * hl) * 72 + t * 16 + ln] = f2bf(pv);   // D-layout -> LDS
      }
    #pragma unroll
    for (int mask = 1; mask < 16; mask <<= 1)
      #pragma unroll
      for (int i = 0; i < 8; i++) rs[i] += __shfl_xor(rs[i], mask, 32);
    #pragma unroll
    for (int i = 0; i < 8; i++) lrow[i] = lrow[i] * alpha[i] + rs[i];
    #pragma unroll
    for (int nt = 0; nt < 4; nt++)
      #pragma unroll
      for (int i = 0; i < 8; i++) o[nt][i] *= alpha[i];

    asm volatile("s_wait_dscnt 0" ::: "memory");  // P visible to our own reads

    // O += P * V  (A-frags of P from LDS, B-frags of V row-major from LDS)
    #pragma unroll
    for (int c = 0; c < 2; c++) {
      Frag fp;
      const u16* p = &Ps[w][ln * 72 + c * 32 + hl * 8];
      fp.q[0] = *(const u32x4*)p; fp.q[1] = *(const u32x4*)(p + 16);
      #pragma unroll
      for (int nt = 0; nt < 4; nt++) {
        Frag fv;
        const u16* pv2 = &Vs[(c * 32) * 72 + nt * 16 + ln];
        #pragma unroll
        for (int s = 0; s < 16; s++) ((u16*)&fv)[s] = pv2[((s >> 3) * 16 + hl * 8 + (s & 7)) * 72];
        o[nt] = __builtin_amdgcn_wmma_f32_16x16x32_bf16(
            false, fp.v, false, fv.v, (short)0, o[nt], false, false);
      }
    }
  }

  float inv[8];
  #pragma unroll
  for (int i = 0; i < 8; i++) inv[i] = 1.0f / lrow[i];
  const int Dm = H * 64;
  #pragma unroll
  for (int nt = 0; nt < 4; nt++)
    #pragma unroll
    for (int i = 0; i < 8; i++) {
      int r = qbase + w * 16 + i + hl * 8;
      int c = h * 64 + nt * 16 + ln;
      O[((size_t)b * S + r) * Dm + c] = f2bf(o[nt][i] * inv[i]);
    }
}

extern "C" void kernel_launch(void* const* d_in, const int* in_sizes, int n_in,
                              void* d_out, int out_size, void* d_ws, size_t ws_size,
                              hipStream_t stream) {
  const float* x  = (const float*)d_in[0];
  const float* Wq = (const float*)d_in[1];
  const float* Wk = (const float*)d_in[2];
  const float* Wv = (const float*)d_in[3];
  const float* Wo = (const float*)d_in[4];
  const float* bo = (const float*)d_in[5];
  const int B = 4, S = 2048, D = 1024, H = 16;
  const size_t nx = (size_t)B * S * D;   // 8,388,608
  const size_t nw = (size_t)D * D;       // 1,048,576

  u16* ws  = (u16*)d_ws;
  u16* wqb = ws;
  u16* wkb = wqb + nw;
  u16* wvb = wkb + nw;
  u16* wob = wvb + nw;
  u16* Qb  = wob + nw;
  u16* Kb  = Qb + nx;
  u16* Vb  = Kb + nx;
  u16* AO  = Vb + nx;                    // total ~75.5 MB of workspace

  cvt_bf16<<<1024, 256, 0, stream>>>(Wq, wqb, (int)nw);
  cvt_bf16<<<1024, 256, 0, stream>>>(Wk, wkb, (int)nw);
  cvt_bf16<<<1024, 256, 0, stream>>>(Wv, wvb, (int)nw);
  cvt_bf16<<<1024, 256, 0, stream>>>(Wo, wob, (int)nw);

  dim3 gg(D / 128, (B * S) / 128);       // 8 x 64 blocks
  gemm128_bf16<<<gg, 256, 0, stream>>>(x, 1, wqb, B * S, D, D, S, H,
                                       Qb, nullptr, nullptr, 0);
  gemm128_bf16<<<gg, 256, 0, stream>>>(x, 1, wkb, B * S, D, D, S, H,
                                       Kb, nullptr, nullptr, 0);
  gemm128_bf16<<<gg, 256, 0, stream>>>(x, 1, wvb, B * S, D, D, S, H,
                                       Vb, nullptr, nullptr, 0);

  attn64<<<dim3(S / 64, H, B), 128, 0, stream>>>(Qb, Kb, Vb, AO, S, H);

  gemm128_bf16<<<gg, 256, 0, stream>>>(AO, 0, wob, B * S, D, D, S, H,
                                       nullptr, (float*)d_out, bo, 1);
}
